// Qatten_Weight_69655779606709
// MI455X (gfx1250) — matrix operand order repaired and checked by reading, not compiled, not run on hardware
//
#include <hip/hip_runtime.h>

#define BB 16384
#define AA 8
#define SS 512
#define UU 64
#define HH 4
#define EE 64
#define HEHE 256

// bf16 weight workspace layout (element offsets in d_ws)
#define W1T_OFF 0u          // [h][f=256][k=512]  <- sel_w1[h][k][f]
#define W2T_OFF 524288u     // [h][e=64][k=256]   <- sel_w2[h][k][e]
#define KWT_OFF 589824u     // [h][u=64][e=64]    <- key_w (straight)
#define VW1T_OFF 606208u    // [n=64][k=512]      <- v_w1[k][n]
#define WS_TOTAL 638976u

#define SLAB_RS 40          // bslab row stride (elems): 80B -> 16B-aligned, conflict-free

typedef __bf16 bf16;
typedef __attribute__((ext_vector_type(16))) bf16 v16bf;
typedef __attribute__((ext_vector_type(4))) bf16 v4bf;
typedef __attribute__((ext_vector_type(8))) float v8f;

// A fragment: 16x32 bf16 tile, row-major LDS source with row stride rs.
// ISA layout: lanes 0-15 hold K={0..7,16..23}, lanes 16-31 hold K={8..15,24..31}.
__device__ inline v16bf load_afrag(const bf16* base, int rs, int kofs) {
  int lane = threadIdx.x & 31;
  int m = lane & 15;
  int kh = (lane >> 4) << 3;  // 0 or 8
  const bf16* p = base + m * rs + kofs + kh;
  v16bf a;
#pragma unroll
  for (int i = 0; i < 8; ++i) { a[i] = p[i]; a[8 + i] = p[16 + i]; }
  return a;
}

// B fragment: 32x16 tile from K-transposed slab bslabT[n][k], row stride SLAB_RS.
// ISA layout: lanes 0-15 hold K={0..7,16..23} of column n=lane, lanes 16-31 hold K={8..15,24..31}.
__device__ inline v16bf load_bfrag(const bf16* slab, int ntile) {
  int lane = threadIdx.x & 31;
  const bf16* p = slab + (ntile * 16 + (lane & 15)) * SLAB_RS + ((lane >> 4) << 3);
  v16bf b;
#pragma unroll
  for (int i = 0; i < 8; ++i) { b[i] = p[i]; b[8 + i] = p[16 + i]; }
  return b;
}

// gfx1250 async global->LDS copy of 16 bytes (GVS mode: SGPR base + u32 offset).
// Tracked by ASYNCcnt.
__device__ inline void async_b128(unsigned int ldsOff, const bf16* base,
                                  unsigned int elemOff) {
  asm volatile("global_load_async_to_lds_b128 %0, %1, %2"
               :: "v"(ldsOff), "v"(elemOff * 2u), "s"(base)
               : "memory");
}
__device__ inline void wait_async0() {
  asm volatile("s_wait_asynccnt 0" ::: "memory");
}

__global__ __launch_bounds__(256)
void convert_weights_kernel(const float* __restrict__ sel_w1,
                            const float* __restrict__ sel_w2,
                            const float* __restrict__ key_w,
                            const float* __restrict__ v_w1,
                            bf16* __restrict__ ws) {
  unsigned int gid = blockIdx.x * 256 + threadIdx.x;
  if (gid < W2T_OFF) {                           // W1T
    int k = gid & 511, f = (gid >> 9) & 255, h = gid >> 17;
    ws[gid] = (bf16)sel_w1[(h * SS + k) * HEHE + f];
  } else if (gid < KWT_OFF) {                    // W2T
    unsigned int i = gid - W2T_OFF;
    int k = i & 255, e = (i >> 8) & 63, h = i >> 14;
    ws[gid] = (bf16)sel_w2[(h * HEHE + k) * EE + e];
  } else if (gid < VW1T_OFF) {                   // KWT (straight copy)
    ws[gid] = (bf16)key_w[gid - KWT_OFF];
  } else if (gid < WS_TOTAL) {                   // VW1T
    unsigned int i = gid - VW1T_OFF;
    int k = i & 511, n = i >> 9;
    ws[gid] = (bf16)v_w1[k * EE + n];
  }
}

__global__ void zero_tail_kernel(float* __restrict__ out) {
  if (threadIdx.x < 5) out[(size_t)BB * AA + BB + threadIdx.x] = 0.f;
}

__global__ __launch_bounds__(256)
void qatten_kernel(const float* __restrict__ states,
                   const float* __restrict__ sel_b1,
                   const float* __restrict__ v_b1,
                   const float* __restrict__ v_w2,
                   const float* __restrict__ v_b2,
                   const bf16* __restrict__ wsb,
                   float* __restrict__ out) {
  __shared__ bf16 stA[32 * 520];                 // st tile, bf16
  __shared__ bf16 bslab[64 * SLAB_RS];           // K-transposed B slab (async dst)
  __shared__ __align__(16) unsigned char hmidRaw[32 * 264 * 2];
  __shared__ bf16 selA[32 * 72];
  __shared__ float logitsBuf[32 * 4 * 8];
  __shared__ float magAcc;
  __shared__ float entAcc[4];

  bf16* hmidA = (bf16*)hmidRaw;    // [32][264] bf16
  bf16* tb1   = (bf16*)hmidRaw;    // overlap: [32][66] bf16 (hmid dead)
  float* vact = (float*)hmidRaw;   // overlap: [32][66] f32  (hmid dead)

  const int tid  = threadIdx.x;
  const int lane = tid & 31;
  const int wave = tid >> 5;
  const int mtile = wave & 1;
  const int ntile = (wave >> 1) & 3;
  const int b0 = blockIdx.x * 32;

  // per-thread async slab destination (one b128 per thread, 16B aligned)
  const int sn = tid >> 2;                 // 0..63  (slab row)
  const int skq = (tid & 3) * 8;           // 0,8,16,24 (k quarter)
  const unsigned int slabLds =
      (unsigned int)(uintptr_t)&bslab[sn * SLAB_RS + skq];

  if (tid == 0) magAcc = 0.f;
  if (tid < 4) entAcc[tid] = 0.f;

  // Stage st = states[:, 0, :] as bf16 (float4 loads, packed bf16 stores).
#pragma unroll 4
  for (int j = 0; j < 16; ++j) {
    int i = tid + j * 256;                 // vec4 index, 4096 total
    int row = i >> 7;
    int s4 = (i & 127) << 2;
    float4 f = *(const float4*)&states[(size_t)(b0 + row) * (AA * SS) + s4];
    v4bf c;
    c[0] = (bf16)f.x; c[1] = (bf16)f.y; c[2] = (bf16)f.z; c[3] = (bf16)f.w;
    *(v4bf*)&stA[row * 520 + s4] = c;
  }
  __syncthreads();

  float magPart = 0.f;

  for (int h = 0; h < HH; ++h) {
    // --- GEMM1: hmid_h = relu(st @ W1_h + b1_h), 256 feats = 4 ngroups of 64
    for (int ng = 0; ng < 4; ++ng) {
      v8f acc = {};
      for (int ks = 0; ks < 16; ++ks) {
        __syncthreads();
        async_b128(slabLds, wsb,
                   W1T_OFF + (unsigned)(h * 256 + ng * 64 + sn) * 512 + ks * 32 + skq);
        wait_async0();
        __syncthreads();
        v16bf a = load_afrag(stA + mtile * 16 * 520, 520, ks * 32);
        v16bf b = load_bfrag(bslab, ntile);
        acc = __builtin_amdgcn_wmma_f32_16x16x32_bf16(false, a, false, b, (short)0, acc, false, false);
      }
      {
        int n = ntile * 16 + (lane & 15);
        int fl = ng * 64 + n;
        float bias = sel_b1[h * HEHE + fl];
#pragma unroll
        for (int vv = 0; vv < 8; ++vv) {
          int m = mtile * 16 + vv + ((lane >> 4) << 3);
          float val = acc[vv] + bias;
          hmidA[m * 264 + fl] = (bf16)(val > 0.f ? val : 0.f);
        }
      }
    }
    // --- GEMM2: sel_h = hmid_h @ W2_h  (N=64, K=256)
    {
      v8f acc = {};
      for (int ks = 0; ks < 8; ++ks) {
        __syncthreads();
        async_b128(slabLds, wsb,
                   W2T_OFF + (unsigned)(h * 64 + sn) * 256 + ks * 32 + skq);
        wait_async0();
        __syncthreads();
        v16bf a = load_afrag(hmidA + mtile * 16 * 264, 264, ks * 32);
        v16bf b = load_bfrag(bslab, ntile);
        acc = __builtin_amdgcn_wmma_f32_16x16x32_bf16(false, a, false, b, (short)0, acc, false, false);
      }
      int n = ntile * 16 + (lane & 15);
#pragma unroll
      for (int vv = 0; vv < 8; ++vv) {
        int m = mtile * 16 + vv + ((lane >> 4) << 3);
        selA[m * 72 + n] = (bf16)acc[vv];
      }
    }
    // --- GEMM_T: t_h = sel_h @ key_w_h^T  (N=64 u, K=64 e)
    {
      v8f acc = {};
      for (int ks = 0; ks < 2; ++ks) {
        __syncthreads();
        async_b128(slabLds, wsb,
                   KWT_OFF + (unsigned)(h * 64 + sn) * 64 + ks * 32 + skq);
        wait_async0();
        __syncthreads();
        v16bf a = load_afrag(selA + mtile * 16 * 72, 72, ks * 32);
        v16bf b = load_bfrag(bslab, ntile);
        acc = __builtin_amdgcn_wmma_f32_16x16x32_bf16(false, a, false, b, (short)0, acc, false, false);
      }
      int n = ntile * 16 + (lane & 15);
#pragma unroll
      for (int vv = 0; vv < 8; ++vv) {
        int m = mtile * 16 + vv + ((lane >> 4) << 3);
        tb1[m * 66 + n] = (bf16)acc[vv];
      }
    }
    __syncthreads();
    // --- logits[b,h,a] = st[b, a*64:(a+1)*64] . t[b,h,:]
    {
      int row = tid >> 3, a = tid & 7;
      float s = 0.f;
      const bf16* su = stA + row * 520 + a * 64;
      const bf16* tu = tb1 + row * 66;
#pragma unroll 16
      for (int u = 0; u < 64; ++u) s += (float)su[u] * (float)tu[u];
      logitsBuf[(row * 4 + h) * 8 + a] = s;
      magPart += s * s;
    }
    __syncthreads();
  }

  atomicAdd(&magAcc, magPart);

  // --- softmax over agents, entropy per head
  if (tid < 128) {
    int row = tid >> 2, h = tid & 3;
    float* Lp = logitsBuf + (row * 4 + h) * 8;
    float l[8], mx = -3.4e38f;
#pragma unroll
    for (int a = 0; a < 8; ++a) { l[a] = Lp[a] * 0.125f; mx = fmaxf(mx, l[a]); }
    float sum = 0.f;
#pragma unroll
    for (int a = 0; a < 8; ++a) { l[a] = __expf(l[a] - mx); sum += l[a]; }
    float inv = 1.f / sum, ent = 0.f;
#pragma unroll
    for (int a = 0; a < 8; ++a) {
      float w = l[a] * inv;
      Lp[a] = w;
      ent += w * __logf(w + 1e-8f);
    }
    atomicAdd(&entAcc[h], ent);
  }
  __syncthreads();

  // --- head_attend = sum over heads of weights
  {
    int row = tid >> 3, a = tid & 7;
    float s = 0.f;
#pragma unroll
    for (int h = 0; h < 4; ++h) s += logitsBuf[(row * 4 + h) * 8 + a];
    out[(size_t)(b0 + row) * AA + a] = s;
  }
  if (tid == 0)
    atomicAdd(out + (size_t)BB * AA + BB, magAcc * (0.001f / ((float)BB * (float)AA)));
  if (tid < 4)
    atomicAdd(out + (size_t)BB * AA + BB + 1 + tid, -entAcc[tid] / (float)BB);

  // --- V net: relu(st @ v_w1 + v_b1) @ v_w2 + v_b2
  {
    v8f acc = {};
    for (int ks = 0; ks < 16; ++ks) {
      __syncthreads();
      async_b128(slabLds, wsb,
                 VW1T_OFF + (unsigned)sn * 512 + ks * 32 + skq);
      wait_async0();
      __syncthreads();
      v16bf a = load_afrag(stA + mtile * 16 * 520, 520, ks * 32);
      v16bf b = load_bfrag(bslab, ntile);
      acc = __builtin_amdgcn_wmma_f32_16x16x32_bf16(false, a, false, b, (short)0, acc, false, false);
    }
    int n = ntile * 16 + (lane & 15);
    float bias = v_b1[n];
#pragma unroll
    for (int vv = 0; vv < 8; ++vv) {
      int m = mtile * 16 + vv + ((lane >> 4) << 3);
      float val = acc[vv] + bias;
      vact[m * 66 + n] = val > 0.f ? val : 0.f;
    }
  }
  __syncthreads();
  if (tid < 32) {
    int row = tid;
    float s = v_b2[0];
    const float* vr = vact + row * 66;
#pragma unroll 16
    for (int nn = 0; nn < 64; ++nn) s += vr[nn] * v_w2[nn];
    out[(size_t)BB * AA + b0 + row] = s;
  }
}

extern "C" void kernel_launch(void* const* d_in, const int* in_sizes, int n_in,
                              void* d_out, int out_size, void* d_ws, size_t ws_size,
                              hipStream_t stream) {
  (void)in_sizes; (void)n_in; (void)out_size; (void)ws_size;
  const float* states = (const float*)d_in[1];
  const float* sel_w1 = (const float*)d_in[3];
  const float* sel_b1 = (const float*)d_in[4];
  const float* sel_w2 = (const float*)d_in[5];
  const float* key_w  = (const float*)d_in[6];
  const float* v_w1   = (const float*)d_in[7];
  const float* v_b1   = (const float*)d_in[8];
  const float* v_w2   = (const float*)d_in[9];
  const float* v_b2   = (const float*)d_in[10];
  float* out = (float*)d_out;
  bf16* wsb = (bf16*)d_ws;

  hipLaunchKernelGGL(convert_weights_kernel, dim3((WS_TOTAL + 255) / 256), dim3(256),
                     0, stream, sel_w1, sel_w2, key_w, v_w1, wsb);
  hipLaunchKernelGGL(zero_tail_kernel, dim3(1), dim3(32), 0, stream, out);
  hipLaunchKernelGGL(qatten_kernel, dim3(BB / 32), dim3(256), 0, stream,
                     states, sel_b1, v_b1, v_w2, v_b2, wsb, out);
}
